// KFGN_3977139716602
// MI455X (gfx1250) — compile-verified, matrix operand order checked by reading, not run on hardware
//
#include <hip/hip_runtime.h>
#include <hip/hip_bf16.h>

// ---------------------------------------------------------------------------
// CDNA5 (gfx1250) implementation. Wave32.
//   - Large GEMMs: v_wmma_f32_16x16x32_bf16, bf16 operands, f32 accumulate.
//   - Tile staging: global_load_async_to_lds_b128 (ASYNCcnt) + double buffering.
// ---------------------------------------------------------------------------

#define USE_ASYNC_LDS 1

typedef __attribute__((ext_vector_type(16))) __bf16 v16bf;
typedef __attribute__((ext_vector_type(8)))  float  v8f;

#define F_DIM 512
#define B_DIM 64
#define T_DIM 256

__device__ __forceinline__ unsigned short f2bf(float f) {
  unsigned int u = __float_as_uint(f);
  unsigned int r = u + 0x7FFFu + ((u >> 16) & 1u);   // round-to-nearest-even
  return (unsigned short)(r >> 16);
}

__device__ __forceinline__ float bf2f(unsigned short h) {
  return __uint_as_float(((unsigned int)h) << 16);
}

__device__ __forceinline__ float sigmoidf_(float x) {
  return 1.0f / (1.0f + __expf(-x));
}

// ---------------------------------------------------------------------------
// Small fp32 kernels (A normalization / powers / W_k build) — negligible FLOPs
// ---------------------------------------------------------------------------

__global__ __launch_bounds__(256) void colsum_kernel(const float* __restrict__ A,
                                                     float* __restrict__ s) {
  int j = blockIdx.x * 256 + threadIdx.x;       // 512 columns
  float acc = 0.f;
  for (int i = 0; i < F_DIM; ++i) acc += A[i * F_DIM + j];
  s[j] = acc;
}

__global__ __launch_bounds__(256) void an_kernel(const float* __restrict__ A,
                                                 const float* __restrict__ s,
                                                 float* __restrict__ An,
                                                 float* __restrict__ A1) {
  int idx = blockIdx.x * 256 + threadIdx.x;     // 512*512
  int i = idx >> 9;                              // row
  float v = A[idx] / s[i];                       // (D_inv @ A)[i,j] = A[i,j]/colsum[i]
  An[idx] = v;
  A1[idx] = fminf(v, 1.0f);
}

// Y = min(X @ An, 1.0), 512x512x512 fp32 tiled
__global__ __launch_bounds__(256) void powmat_kernel(const float* __restrict__ X,
                                                     const float* __restrict__ An,
                                                     float* __restrict__ Y) {
  __shared__ float sx[16][17], sa[16][17];
  int tx = threadIdx.x & 15, ty = threadIdx.x >> 4;
  int row = blockIdx.y * 16 + ty, col = blockIdx.x * 16 + tx;
  float acc = 0.f;
  for (int k0 = 0; k0 < F_DIM; k0 += 16) {
    sx[ty][tx] = X[row * F_DIM + k0 + tx];
    sa[ty][tx] = An[(k0 + ty) * F_DIM + col];
    __syncthreads();
#pragma unroll
    for (int kk = 0; kk < 16; ++kk) acc += sx[ty][kk] * sa[kk][tx];
    __syncthreads();
  }
  Y[row * F_DIM + col] = fminf(acc, 1.0f);
}

// Wk[g,f] = bf16( Ak[g,f] * sum_j gcw[g,j]*gct[f,j] )  (weight layout [N=g, K=f])
__global__ __launch_bounds__(256) void wkbuild_kernel(const float* __restrict__ Gw,
                                                      const float* __restrict__ Gt,
                                                      const float* __restrict__ Ak,
                                                      unsigned short* __restrict__ Wk) {
  __shared__ float sg[16][17], st[16][17];
  int tx = threadIdx.x & 15, ty = threadIdx.x >> 4;
  int g = blockIdx.y * 16 + ty;
  int f0 = blockIdx.x * 16;
  float acc = 0.f;
  for (int k0 = 0; k0 < F_DIM; k0 += 16) {
    sg[ty][tx] = Gw[g * F_DIM + k0 + tx];
    st[ty][tx] = Gt[(f0 + ty) * F_DIM + k0 + tx];
    __syncthreads();
#pragma unroll
    for (int kk = 0; kk < 16; ++kk) acc += sg[ty][kk] * st[tx][kk];
    __syncthreads();
  }
  int f = f0 + tx;
  Wk[g * F_DIM + f] = f2bf(Ak[g * F_DIM + f] * acc);
}

__global__ __launch_bounds__(256) void cvt_bf16_kernel(const float* __restrict__ src,
                                                       unsigned short* __restrict__ dst,
                                                       int n) {
  int idx = blockIdx.x * 256 + threadIdx.x;
  if (idx < n) dst[idx] = f2bf(src[idx]);
}

// ---------------------------------------------------------------------------
// WMMA GEMM: out[m,n] = epilogue( sum_k A_bf16[m,k] * B_bf16[n,k] )
//   Block = 128 threads (4 waves), tile BM=64 x BN=64 x BK=32, double-buffered.
//   Staging via global_load_async_to_lds_b128 (ASYNCcnt) -> LDS, overlapped
//   with WMMA on the other buffer.
//   NB==1 : bf16 store with affine row remap (gc layout [B,3T,F]).
//   NB==3 : fused gate epilogue  out = sig(z1+b1)*tanh(sig(z0+b0)*tanh(z2+b2)).
// ---------------------------------------------------------------------------
template <int NB>
__global__ __launch_bounds__(128) void wmma_gemm_kernel(
    const unsigned short* __restrict__ Ab, int lda,
    const unsigned short* __restrict__ B0,
    const unsigned short* __restrict__ B1,
    const unsigned short* __restrict__ B2,
    const float* __restrict__ bias0,
    const float* __restrict__ bias1,
    const float* __restrict__ bias2,
    float* __restrict__ out, unsigned short* __restrict__ outb, int ldo,
    int K, int rsh, int rmul, int radd) {
  // [2 buffers][A tile + NB B tiles], each tile 64x32 bf16 = 4KB
  __shared__ __attribute__((aligned(16))) unsigned short lds[2 * 2048 * (1 + NB)];

  const int tid  = threadIdx.x;
  const int wave = tid >> 5;
  const int lane = tid & 31;
  const int half = lane >> 4;
  const int r    = lane & 15;
  const int m0   = blockIdx.y * 64;
  const int n0   = blockIdx.x * 64;

  v8f acc[NB][4];
#pragma unroll
  for (int b = 0; b < NB; ++b)
#pragma unroll
    for (int j = 0; j < 4; ++j) acc[b][j] = {};

  // Issue async copies for one BK=32 stage into LDS buffer `buf`.
  auto issue = [&](int buf, int k0) {
    const int base = buf * 2048 * (1 + NB);
    // A tile: 64 rows x 32 bf16 = 256 x 16B, 2 per thread
#pragma unroll
    for (int it = 0; it < 2; ++it) {
      int idx = it * 128 + tid;
      int row = idx >> 2, cc = idx & 3;
      const unsigned short* g = Ab + (size_t)(m0 + row) * lda + k0 + cc * 8;
      unsigned int lo = (unsigned int)(size_t)&lds[base + row * 32 + cc * 8];
#if USE_ASYNC_LDS
      asm volatile("global_load_async_to_lds_b128 %0, %1, off"
                   :: "v"(lo), "v"(g) : "memory");
#else
      *(uint4*)&lds[base + row * 32 + cc * 8] = *(const uint4*)g;
      (void)lo;
#endif
    }
    // B tiles
#pragma unroll
    for (int b = 0; b < NB; ++b) {
      const unsigned short* Bp = (b == 0 ? B0 : (b == 1 ? B1 : B2));
#pragma unroll
      for (int it = 0; it < 2; ++it) {
        int idx = it * 128 + tid;
        int row = idx >> 2, cc = idx & 3;
        const unsigned short* g = Bp + (size_t)(n0 + row) * K + k0 + cc * 8;
        unsigned int lo =
            (unsigned int)(size_t)&lds[base + 2048 * (1 + b) + row * 32 + cc * 8];
#if USE_ASYNC_LDS
        asm volatile("global_load_async_to_lds_b128 %0, %1, off"
                     :: "v"(lo), "v"(g) : "memory");
#else
        *(uint4*)&lds[base + 2048 * (1 + b) + row * 32 + cc * 8] = *(const uint4*)g;
        (void)lo;
#endif
      }
    }
  };

  const int nk = K >> 5;
  issue(0, 0);

  for (int kk = 0; kk < nk; ++kk) {
#if USE_ASYNC_LDS
    asm volatile("s_wait_asynccnt 0x0" ::: "memory");  // own copies done
#endif
    __syncthreads();  // all waves' copies done; prev compute done everywhere
    if (kk + 1 < nk) issue((kk + 1) & 1, (kk + 1) * 32);  // overlap with WMMA

    const int base = (kk & 1) * 2048 * (1 + NB);
    union Frag { v16bf v; uint4 q[2]; } af, bfr;
    {
      // A frag: row M = lane%16 of this wave's strip; K = [8h,8h+8) U [16+8h,+8)
      const unsigned short* ar = &lds[base + (wave * 16 + r) * 32];
      af.q[0] = *(const uint4*)(ar + 8 * half);
      af.q[1] = *(const uint4*)(ar + 16 + 8 * half);
    }
#pragma unroll
    for (int b = 0; b < NB; ++b) {
#pragma unroll
      for (int j = 0; j < 4; ++j) {
        const unsigned short* br = &lds[base + 2048 * (1 + b) + (j * 16 + r) * 32];
        bfr.q[0] = *(const uint4*)(br + 8 * half);
        bfr.q[1] = *(const uint4*)(br + 16 + 8 * half);
        acc[b][j] = __builtin_amdgcn_wmma_f32_16x16x32_bf16(
            false, af.v, false, bfr.v, (short)0, acc[b][j], false, false);
      }
    }
    __syncthreads();
  }

  // Epilogue. C frag: VGPR i -> row (i + 8*half), col = lane%16 within 16x16 tile.
  const int mbase = m0 + wave * 16;
#pragma unroll
  for (int j = 0; j < 4; ++j) {
    const int col = n0 + j * 16 + r;
#pragma unroll
    for (int i = 0; i < 8; ++i) {
      const int m    = mbase + i + 8 * half;
      const int orow = m + ((m >> rsh) * rmul) + radd;
      if (NB == 1) {
        outb[(size_t)orow * ldo + col] = f2bf(acc[0][j][i]);
      } else {
        const float zi = acc[0][j][i] + bias0[col];
        const float zo = acc[1][j][i] + bias1[col];
        const float zc = acc[2][j][i] + bias2[col];
        out[(size_t)orow * ldo + col] =
            sigmoidf_(zo) * tanhf(sigmoidf_(zi) * tanhf(zc));
      }
    }
  }
}

// ---------------------------------------------------------------------------
// Deterministic two-stage variance reduction (sum, sumsq)
// ---------------------------------------------------------------------------
__global__ __launch_bounds__(256) void reduce1_f32_kernel(const float* __restrict__ x,
                                                          int n, float* __restrict__ part) {
  __shared__ float ss[256], sq[256];
  float s = 0.f, q = 0.f;
  for (int i = blockIdx.x * 256 + threadIdx.x; i < n; i += gridDim.x * 256) {
    float v = x[i]; s += v; q += v * v;
  }
  ss[threadIdx.x] = s; sq[threadIdx.x] = q;
  __syncthreads();
  for (int o = 128; o > 0; o >>= 1) {
    if (threadIdx.x < o) {
      ss[threadIdx.x] += ss[threadIdx.x + o];
      sq[threadIdx.x] += sq[threadIdx.x + o];
    }
    __syncthreads();
  }
  if (threadIdx.x == 0) { part[blockIdx.x * 2] = ss[0]; part[blockIdx.x * 2 + 1] = sq[0]; }
}

__global__ __launch_bounds__(256) void reduce1_bf16_kernel(
    const unsigned short* __restrict__ x, int n, float* __restrict__ part) {
  __shared__ float ss[256], sq[256];
  float s = 0.f, q = 0.f;
  for (int i = blockIdx.x * 256 + threadIdx.x; i < n; i += gridDim.x * 256) {
    float v = bf2f(x[i]); s += v; q += v * v;
  }
  ss[threadIdx.x] = s; sq[threadIdx.x] = q;
  __syncthreads();
  for (int o = 128; o > 0; o >>= 1) {
    if (threadIdx.x < o) {
      ss[threadIdx.x] += ss[threadIdx.x + o];
      sq[threadIdx.x] += sq[threadIdx.x + o];
    }
    __syncthreads();
  }
  if (threadIdx.x == 0) { part[blockIdx.x * 2] = ss[0]; part[blockIdx.x * 2 + 1] = sq[0]; }
}

__global__ __launch_bounds__(256) void reduce2_kernel(const float* __restrict__ part,
                                                      int nb, float n,
                                                      float* __restrict__ var_out) {
  __shared__ float ss[256], sq[256];
  float s = 0.f, q = 0.f;
  for (int i = threadIdx.x; i < nb; i += 256) { s += part[i * 2]; q += part[i * 2 + 1]; }
  ss[threadIdx.x] = s; sq[threadIdx.x] = q;
  __syncthreads();
  for (int o = 128; o > 0; o >>= 1) {
    if (threadIdx.x < o) {
      ss[threadIdx.x] += ss[threadIdx.x + o];
      sq[threadIdx.x] += sq[threadIdx.x + o];
    }
    __syncthreads();
  }
  if (threadIdx.x == 0)
    var_out[0] = (sq[0] - ss[0] * ss[0] / n) / (n - 1.0f);  // ddof=1
}

// ---------------------------------------------------------------------------
// Final blend. Rows with all-zero "combined" inputs collapse to bias-only
// constants, synthesized inline.
// ---------------------------------------------------------------------------
__global__ __launch_bounds__(256) void final_kernel(
    const float* __restrict__ H,  const float* __restrict__ R,
    const float* __restrict__ bi, const float* __restrict__ bo, const float* __restrict__ bc,
    const float* __restrict__ rbi, const float* __restrict__ rbo, const float* __restrict__ rbc,
    const float* __restrict__ vars, const float* __restrict__ cptr,
    float* __restrict__ out) {
  int idx = blockIdx.x * 256 + threadIdx.x;       // B*T*F = 8388608
  int g = idx & (F_DIM - 1);
  int t = (idx >> 9) & (T_DIM - 1);
  int b = idx >> 17;
  const float var1 = vars[0], var2 = vars[1], c = cptr[0];

  float hid;
  if (t < 192) hid = H[((size_t)(b * 192 + t) << 9) + g];
  else         hid = sigmoidf_(bo[g]) * tanhf(sigmoidf_(bi[g]) * tanhf(bc[g]));

  float rhid;
  if (t < 128) rhid = R[((size_t)(b * 128 + t) << 9) + g];
  else         rhid = sigmoidf_(rbo[g]) * tanhf(sigmoidf_(rbi[g]) * tanhf(rbc[g]));

  out[idx] = (hid * var1 * c + rhid * var2) / (var1 + var2 * c);
}

// ---------------------------------------------------------------------------
// Host side
// ---------------------------------------------------------------------------
static inline char* carve(char*& p, size_t bytes) {
  char* r = p;
  p += (bytes + 255) & ~(size_t)255;
  return r;
}

extern "C" void kernel_launch(void* const* d_in, const int* in_sizes, int n_in,
                              void* d_out, int out_size, void* d_ws, size_t ws_size,
                              hipStream_t stream) {
  (void)in_sizes; (void)n_in; (void)out_size; (void)ws_size;
  const float* input = (const float*)d_in[0];
  const float* A     = (const float*)d_in[1];
  const float* gcw   = (const float*)d_in[2];
  const float* gct   = (const float*)d_in[3];
  // d_in[4]=Wf, d_in[5]=bf : dead (multiplied by zero initial Cell)
  const float* Wi  = (const float*)d_in[6];  const float* bi  = (const float*)d_in[7];
  const float* Wo  = (const float*)d_in[8];  const float* bo  = (const float*)d_in[9];
  const float* Wc  = (const float*)d_in[10]; const float* bc  = (const float*)d_in[11];
  // d_in[12]=rWf, d_in[13]=rbf : dead
  const float* rWi = (const float*)d_in[14]; const float* rbi = (const float*)d_in[15];
  const float* rWo = (const float*)d_in[16]; const float* rbo = (const float*)d_in[17];
  const float* rWc = (const float*)d_in[18]; const float* rbc = (const float*)d_in[19];
  // d_in[20]=neighbor_weight : dead
  const float* cs  = (const float*)d_in[21];

  const size_t FF = (size_t)F_DIM * F_DIM;
  const int NTOT = B_DIM * T_DIM * F_DIM;           // 8388608

  char* p = (char*)d_ws;
  float*          s_col  = (float*)carve(p, F_DIM * 4);
  float*          An     = (float*)carve(p, FF * 4);
  float*          A1f    = (float*)carve(p, FF * 4);
  float*          A2f    = (float*)carve(p, FF * 4);
  float*          A3f    = (float*)carve(p, FF * 4);
  unsigned short* Wkb    = (unsigned short*)carve(p, 3 * FF * 2);
  unsigned short* Wib    = (unsigned short*)carve(p, (size_t)F_DIM * 2048 * 2);
  unsigned short* Wob    = (unsigned short*)carve(p, (size_t)F_DIM * 2048 * 2);
  unsigned short* Wcb    = (unsigned short*)carve(p, (size_t)F_DIM * 2048 * 2);
  unsigned short* rWib   = (unsigned short*)carve(p, (size_t)F_DIM * 1024 * 2);
  unsigned short* rWob   = (unsigned short*)carve(p, (size_t)F_DIM * 1024 * 2);
  unsigned short* rWcb   = (unsigned short*)carve(p, (size_t)F_DIM * 1024 * 2);
  unsigned short* inputb = (unsigned short*)carve(p, (size_t)NTOT * 2);          // 16.8 MB
  unsigned short* gcb    = (unsigned short*)carve(p, (size_t)B_DIM * 768 * F_DIM * 2); // 50.3 MB
  float*          Hid    = (float*)carve(p, (size_t)B_DIM * 192 * F_DIM * 4);    // 25.2 MB
  float*          Rhid   = (float*)carve(p, (size_t)B_DIM * 128 * F_DIM * 4);    // 16.8 MB
  float*          part   = (float*)carve(p, 512 * 2 * 4);
  float*          vars   = (float*)carve(p, 2 * 4);

  // 1) An = D_inv @ A ; A1 = min(An,1); A2,A3 cumulative clamped powers
  colsum_kernel<<<2, 256, 0, stream>>>(A, s_col);
  an_kernel<<<(int)(FF / 256), 256, 0, stream>>>(A, s_col, An, A1f);
  powmat_kernel<<<dim3(32, 32), 256, 0, stream>>>(A1f, An, A2f);
  powmat_kernel<<<dim3(32, 32), 256, 0, stream>>>(A2f, An, A3f);

  // 2) W_k = A_k * (gcw_k @ gct_k^T)  -> bf16 [N=512, K=512]
  const float* Ak[3] = {A1f, A2f, A3f};
  for (int k = 0; k < 3; ++k)
    wkbuild_kernel<<<dim3(32, 32), 256, 0, stream>>>(gcw + k * FF, gct + k * FF,
                                                     Ak[k], Wkb + k * FF);

  // 3) bf16 conversions: gate weights + input
  cvt_bf16_kernel<<<(int)(F_DIM * 2048 / 256), 256, 0, stream>>>(Wi, Wib, F_DIM * 2048);
  cvt_bf16_kernel<<<(int)(F_DIM * 2048 / 256), 256, 0, stream>>>(Wo, Wob, F_DIM * 2048);
  cvt_bf16_kernel<<<(int)(F_DIM * 2048 / 256), 256, 0, stream>>>(Wc, Wcb, F_DIM * 2048);
  cvt_bf16_kernel<<<(int)(F_DIM * 1024 / 256), 256, 0, stream>>>(rWi, rWib, F_DIM * 1024);
  cvt_bf16_kernel<<<(int)(F_DIM * 1024 / 256), 256, 0, stream>>>(rWo, rWob, F_DIM * 1024);
  cvt_bf16_kernel<<<(int)(F_DIM * 1024 / 256), 256, 0, stream>>>(rWc, rWcb, F_DIM * 1024);
  cvt_bf16_kernel<<<NTOT / 256, 256, 0, stream>>>(input, inputb, NTOT);

  // 4) gc_k = input[16384x512] @ W_k^T -> gcb[b, k*256+t, :] (row remap m -> b*768+k*256+t)
  for (int k = 0; k < 3; ++k)
    wmma_gemm_kernel<1><<<dim3(8, 256), 128, 0, stream>>>(
        inputb, F_DIM, Wkb + k * FF, nullptr, nullptr,
        nullptr, nullptr, nullptr, nullptr, gcb, F_DIM,
        /*K=*/F_DIM, /*rsh=*/8, /*rmul=*/512, /*radd=*/k * 256);

  // 5) variances (deterministic 2-stage)
  reduce1_f32_kernel<<<512, 256, 0, stream>>>(input, NTOT, part);
  reduce2_kernel<<<1, 256, 0, stream>>>(part, 512, (float)NTOT, vars + 0);
  reduce1_bf16_kernel<<<512, 256, 0, stream>>>(gcb, B_DIM * 768 * F_DIM, part);
  reduce2_kernel<<<1, 256, 0, stream>>>(part, 512, (float)(B_DIM * 768 * F_DIM), vars + 1);

  // 6) gates: combined = gcb viewed [B*192, 2048]; fused i/o/c epilogue -> Hidden
  wmma_gemm_kernel<3><<<dim3(8, 192), 128, 0, stream>>>(
      gcb, 2048, Wib, Wob, Wcb, bi, bo, bc, Hid, nullptr, F_DIM,
      /*K=*/2048, /*rsh=*/30, /*rmul=*/0, /*radd=*/0);

  // 7) rgates: rcombined = inputb viewed [B*128, 1024] -> rHidden
  wmma_gemm_kernel<3><<<dim3(8, 128), 128, 0, stream>>>(
      inputb, 1024, rWib, rWob, rWcb, rbi, rbo, rbc, Rhid, nullptr, F_DIM,
      /*K=*/1024, /*rsh=*/30, /*rmul=*/0, /*radd=*/0);

  // 8) final blend
  final_kernel<<<NTOT / 256, 256, 0, stream>>>(
      Hid, Rhid, bi, bo, bc, rbi, rbo, rbc, vars, cs, (float*)d_out);
}
